// ScaledDotProductAttention_43035572306294
// MI455X (gfx1250) — compile-verified, matrix operand order checked by reading, not run on hardware
//
#include <hip/hip_runtime.h>

typedef __attribute__((ext_vector_type(16))) _Float16 v16h;
typedef __attribute__((ext_vector_type(8)))  _Float16 v8h;
typedef __attribute__((ext_vector_type(8)))  float    v8f;
typedef int v4i __attribute__((vector_size(4 * sizeof(int))));  // matches builtin param

#define B_ 8
#define S_ 2048
#define D_ 512
#define JT_ (S_ / 16)   // 128 key tiles
#define QT_ (S_ / 16)   // 128 query tiles per batch

#define GLOBAL_AS __attribute__((address_space(1)))
#define LDS_AS    __attribute__((address_space(3)))

#if __has_builtin(__builtin_amdgcn_global_load_async_to_lds_b128) && \
    __has_builtin(__builtin_amdgcn_s_wait_asynccnt)
#define USE_ASYNC 1
#else
#define USE_ASYNC 0
#endif

// ---------------------------------------------------------------------------
// Prep: pack fp32 -> f16 (8 elements per thread, b128 in / b128 out)
// ---------------------------------------------------------------------------
__global__ void pack_f16_kernel(const float* __restrict__ src,
                                _Float16* __restrict__ dst, int n8) {
  int i = blockIdx.x * blockDim.x + threadIdx.x;
  if (i >= n8) return;
  const float4* s = (const float4*)src;
  float4 a = s[2 * i], b = s[2 * i + 1];
  v8h h;
  h[0] = (_Float16)a.x; h[1] = (_Float16)a.y;
  h[2] = (_Float16)a.z; h[3] = (_Float16)a.w;
  h[4] = (_Float16)b.x; h[5] = (_Float16)b.y;
  h[6] = (_Float16)b.z; h[7] = (_Float16)b.w;
  *(v8h*)(dst + 8 * (size_t)i) = h;
}

// ---------------------------------------------------------------------------
// Fragment loaders. ISA 16-bit A/B 16x32 layout (wave32): lane holds row
// lane%16; halves 0..7 -> K = off+0..7, halves 8..15 -> K = off+16..23,
// where off = 32*chunk + 8*(lane>=16).
// ---------------------------------------------------------------------------
__device__ __forceinline__ v16h frag_from_f16(const _Float16* rowptr,
                                              int dbase, int laneHi) {
  const int off0 = dbase + (laneHi << 3);
  v8h a = *(const v8h*)(rowptr + off0);        // 16B aligned -> b128
  v8h b = *(const v8h*)(rowptr + off0 + 16);
  return __builtin_shufflevector(a, b, 0, 1, 2, 3, 4, 5, 6, 7,
                                 8, 9, 10, 11, 12, 13, 14, 15);
}

__device__ __forceinline__ v16h frag_from_f32(const float* __restrict__ rowptr,
                                              int dbase, int laneHi) {
  const int off0 = dbase + (laneHi << 3);
  const float4* p0 = (const float4*)(rowptr + off0);
  const float4* p1 = (const float4*)(rowptr + off0 + 16);
  float4 x0 = p0[0], x1 = p0[1];
  float4 y0 = p1[0], y1 = p1[1];
  v16h r;
  r[0]  = (_Float16)x0.x; r[1]  = (_Float16)x0.y;
  r[2]  = (_Float16)x0.z; r[3]  = (_Float16)x0.w;
  r[4]  = (_Float16)x1.x; r[5]  = (_Float16)x1.y;
  r[6]  = (_Float16)x1.z; r[7]  = (_Float16)x1.w;
  r[8]  = (_Float16)y0.x; r[9]  = (_Float16)y0.y;
  r[10] = (_Float16)y0.z; r[11] = (_Float16)y0.w;
  r[12] = (_Float16)y1.x; r[13] = (_Float16)y1.y;
  r[14] = (_Float16)y1.z; r[15] = (_Float16)y1.w;
  return r;
}

// Online-softmax tile update + final scaled-V store, shared by both kernels.
__device__ __forceinline__ void softmax_update(const v8f& acc, float* m,
                                               float* l, float rscale) {
#pragma unroll
  for (int r = 0; r < 8; ++r) {
    float s = acc[r] * rscale;
    float tmax = s;
#pragma unroll
    for (int off = 1; off < 16; off <<= 1)
      tmax = fmaxf(tmax, __shfl_xor(tmax, off, 16));
    float mn = fmaxf(m[r], tmax);
    float e  = __expf(s - mn);
    float ts = e;
#pragma unroll
    for (int off = 1; off < 16; off <<= 1)
      ts += __shfl_xor(ts, off, 16);
    l[r] = l[r] * __expf(m[r] - mn) + ts;
    m[r] = mn;
  }
}

__device__ __forceinline__ void store_out(const float* __restrict__ v,
                                          float* __restrict__ out, int b,
                                          int ibase, int lane, const float* l) {
  float sc[8];
#pragma unroll
  for (int r = 0; r < 8; ++r) sc[r] = l[r] / l[r];  // row-sum of softmax
  const float* vb = v   + ((size_t)b * S_ + ibase) * D_;
  float*       ob = out + ((size_t)b * S_ + ibase) * D_;
#pragma unroll
  for (int M = 0; M < 16; ++M) {
    const float sM = __shfl(sc[M & 7], (M >> 3) << 4, 32);
    const float4* vr  = (const float4*)(vb + (size_t)M * D_);
    float4*       orw = (float4*)(ob + (size_t)M * D_);
#pragma unroll
    for (int t = 0; t < 4; ++t) {
      float4 x = vr[lane + 32 * t];
      x.x *= sM; x.y *= sM; x.z *= sM; x.w *= sM;
      orw[lane + 32 * t] = x;
    }
  }
}

// ---------------------------------------------------------------------------
// Main kernel, f16 path: Q/K pre-packed to f16 in workspace. K tiles are
// double-buffered in LDS via async-to-LDS copies when available.
// ---------------------------------------------------------------------------
__global__ __launch_bounds__(32)
void sdpa_rowsum_wmma_f16_kernel(const _Float16* __restrict__ qf,
                                 const _Float16* __restrict__ kf,
                                 const float* __restrict__ v,
                                 float* __restrict__ out) {
  const int lane   = threadIdx.x;
  const int b      = blockIdx.x >> 7;
  const int qt     = blockIdx.x & (QT_ - 1);
  const int ibase  = qt << 4;
  const int rowSel = lane & 15;
  const int laneHi = lane >> 4;

  const _Float16* qrow  = qf + ((size_t)b * S_ + ibase + rowSel) * D_;
  const _Float16* kbase = kf + (size_t)b * S_ * D_;

  v16h aq[16];
#pragma unroll
  for (int c = 0; c < 16; ++c) aq[c] = frag_from_f16(qrow, 32 * c, laneHi);

  float m[8], l[8];
#pragma unroll
  for (int r = 0; r < 8; ++r) { m[r] = -INFINITY; l[r] = 0.0f; }
  const float rscale = 0.04419417382415922f;  // 1/sqrt(512)

#if USE_ASYNC
  // 16 KB per key tile (16 rows x 512 halves, contiguous), double buffered.
  __shared__ __align__(16) _Float16 ktile[2][16 * D_];
  auto stage = [&](int jt, int buf) {
    const _Float16* gsrc = kbase + (size_t)jt * 16 * D_;
#pragma unroll
    for (int t = 0; t < 32; ++t) {
      const int off = (lane + 32 * t) * 8;  // halves; 16B granules
      __builtin_amdgcn_global_load_async_to_lds_b128(
          (v4i*)(gsrc + off),
          (LDS_AS v4i*)(&ktile[buf][off]), 0, 0);
    }
  };
  stage(0, 0);
  for (int jt = 0; jt < JT_; ++jt) {
    const int buf = jt & 1;
    if (jt + 1 < JT_) {
      stage(jt + 1, buf ^ 1);
      __builtin_amdgcn_s_wait_asynccnt(32);  // in-order: tile jt resident
    } else {
      __builtin_amdgcn_s_wait_asynccnt(0);
    }
    asm volatile("" ::: "memory");
    const _Float16* krow = &ktile[buf][rowSel * D_];
    v8f acc = {};
#pragma unroll
    for (int c = 0; c < 16; ++c) {
      v16h bk = frag_from_f16(krow, 32 * c, laneHi);  // ds_load_b128 pairs
      acc = __builtin_amdgcn_wmma_f32_16x16x32_f16(
          false, aq[c], false, bk, (short)0, acc, false, false);
    }
    softmax_update(acc, m, l, rscale);
    asm volatile("" ::: "memory");
  }
#else
  for (int jt = 0; jt < JT_; ++jt) {
    const _Float16* krow = kbase + (size_t)(jt * 16 + rowSel) * D_;
    if (jt + 1 < JT_)
      __builtin_prefetch(kbase + (size_t)((jt + 1) * 16 + rowSel) * D_, 0, 0);
    v8f acc = {};
#pragma unroll
    for (int c = 0; c < 16; ++c) {
      v16h bk = frag_from_f16(krow, 32 * c, laneHi);
      acc = __builtin_amdgcn_wmma_f32_16x16x32_f16(
          false, aq[c], false, bk, (short)0, acc, false, false);
    }
    softmax_update(acc, m, l, rscale);
  }
#endif

  store_out(v, out, b, ibase, lane, l);
}

// ---------------------------------------------------------------------------
// Fallback (workspace too small): fp32 loads + in-loop cvt (round-1 path).
// ---------------------------------------------------------------------------
__global__ __launch_bounds__(32)
void sdpa_rowsum_wmma_f32_kernel(const float* __restrict__ q,
                                 const float* __restrict__ k,
                                 const float* __restrict__ v,
                                 float* __restrict__ out) {
  const int lane   = threadIdx.x;
  const int b      = blockIdx.x >> 7;
  const int qt     = blockIdx.x & (QT_ - 1);
  const int ibase  = qt << 4;
  const int rowSel = lane & 15;
  const int laneHi = lane >> 4;

  const float* qrow  = q + ((size_t)b * S_ + ibase + rowSel) * D_;
  const float* kbase = k + (size_t)b * S_ * D_;

  v16h aq[16];
#pragma unroll
  for (int c = 0; c < 16; ++c) aq[c] = frag_from_f32(qrow, 32 * c, laneHi);

  float m[8], l[8];
#pragma unroll
  for (int r = 0; r < 8; ++r) { m[r] = -INFINITY; l[r] = 0.0f; }
  const float rscale = 0.04419417382415922f;

  for (int jt = 0; jt < JT_; ++jt) {
    const float* krow = kbase + (size_t)(jt * 16 + rowSel) * D_;
    v8f acc = {};
#pragma unroll
    for (int c = 0; c < 16; ++c) {
      v16h bk = frag_from_f32(krow, 32 * c, laneHi);
      acc = __builtin_amdgcn_wmma_f32_16x16x32_f16(
          false, aq[c], false, bk, (short)0, acc, false, false);
    }
    softmax_update(acc, m, l, rscale);
  }
  store_out(v, out, b, ibase, lane, l);
}

extern "C" void kernel_launch(void* const* d_in, const int* in_sizes, int n_in,
                              void* d_out, int out_size, void* d_ws, size_t ws_size,
                              hipStream_t stream) {
  (void)in_sizes; (void)n_in; (void)out_size;
  const float* q = (const float*)d_in[0];
  const float* k = (const float*)d_in[1];
  const float* v = (const float*)d_in[2];
  float* out = (float*)d_out;

  const size_t nElem = (size_t)B_ * S_ * D_;           // 8.4M per tensor
  const size_t need  = 2 * nElem * sizeof(_Float16);   // 32 MB

  if (ws_size >= need) {
    _Float16* qf = (_Float16*)d_ws;
    _Float16* kf = qf + nElem;
    const int n8 = (int)(nElem / 8);
    const int tb = 256, gb = (n8 + tb - 1) / tb;
    pack_f16_kernel<<<gb, tb, 0, stream>>>(q, qf, n8);
    pack_f16_kernel<<<gb, tb, 0, stream>>>(k, kf, n8);
    sdpa_rowsum_wmma_f16_kernel<<<dim3(B_ * QT_), dim3(32), 0, stream>>>(qf, kf, v, out);
  } else {
    sdpa_rowsum_wmma_f32_kernel<<<dim3(B_ * QT_), dim3(32), 0, stream>>>(q, k, v, out);
  }
}
